// multimodal_RGCN_69191923138605
// MI455X (gfx1250) — compile-verified, matrix operand order checked by reading, not run on hardware
//
#include <hip/hip_runtime.h>
#include <hip/hip_bf16.h>

// ---------------- problem constants (match reference) ----------------
#define Nn   50000
#define Ee   800000
#define Dd   128
#define Gg   128
#define NPp  25000
#define Ll   2
#define EPSf 1e-5f
#define SLOPE 0.01f   // jax.nn.leaky_relu default

typedef __attribute__((ext_vector_type(16))) _Float16 v16h;
typedef __attribute__((ext_vector_type(8)))  float    v8f;

union A16 { v16h v; _Float16 h[16]; };
union C8  { v8f  v; float    f[8];  };

__device__ __forceinline__ float lrelu(float x) { return x > 0.0f ? x : SLOPE * x; }

// ---------------- utility ----------------
__global__ void k_zero(float* p, int n) {
    int i = blockIdx.x * blockDim.x + threadIdx.x;
    if (i < n) p[i] = 0.0f;
}

// deg[dst] += 1 per edge
__global__ void k_deg(const int* __restrict__ dst, float* __restrict__ deg) {
    int e = blockIdx.x * blockDim.x + threadIdx.x;
    if (e < Ee) atomicAdd(&deg[dst[e]], 1.0f);
}

// dis = (deg + 1)^-1/2   (self loop included)
__global__ void k_dis(const float* __restrict__ deg, float* __restrict__ dis) {
    int i = blockIdx.x * blockDim.x + threadIdx.x;
    if (i < Nn) dis[i] = rsqrtf(deg[i] + 1.0f);
}

// W16T[n*D + k] = (f16) W[k*D + n]   (transpose so B lanes read contiguous K runs)
__global__ void k_convert_w(const float* __restrict__ W, _Float16* __restrict__ WT) {
    int t = blockIdx.x * blockDim.x + threadIdx.x;
    if (t >= Dd * Dd) return;
    int k = t / Dd, n = t % Dd;
    WT[n * Dd + k] = (_Float16)W[k * Dd + n];
}

// ---------------- WMMA GEMM: XW = In(NxD,f32) @ W16T(DxD f16, pre-transposed) ----------------
// 8 waves per block; whole Wt staged once into LDS; each wave computes a 16x128
// strip: A chunk loaded+converted once per K step, reused by 8 WMMAs (one per
// 16-col tile). 4 K-steps x 8 col tiles = 32 v_wmma per wave.
__global__ void __launch_bounds__(256)
k_gemm(const float* __restrict__ In, const _Float16* __restrict__ WT, float* __restrict__ XW) {
    __shared__ _Float16 Bs[Dd * Dd];                 // 32 KB of 320 KB LDS

    // cooperative fill of B (16-byte chunks)
    {
        const uint4* src = (const uint4*)WT;
        uint4*       dst = (uint4*)Bs;
        const int nvec = (Dd * Dd) * 2 / 16;         // 2048
        for (int i = threadIdx.x; i < nvec; i += 256) dst[i] = src[i];
    }
    __syncthreads();

    const int wave = threadIdx.x >> 5;
    const int lane = threadIdx.x & 31;
    const int hs   = lane >> 4;                      // lane half: 0 or 1
    const int lm   = lane & 15;
    const int mb   = (blockIdx.x * 8 + wave) * 16;   // row tile of 16
    if (mb >= Nn) return;                            // whole-wave guard (EXEC stays all-1s)

    C8 acc[8];
    #pragma unroll
    for (int nt = 0; nt < 8; ++nt)
        #pragma unroll
        for (int r = 0; r < 8; ++r) acc[nt].f[r] = 0.0f;

    const float* arow = In + (size_t)(mb + lm) * Dd + hs * 8;   // A: row mb+lm

    #pragma unroll
    for (int kc = 0; kc < 4; ++kc) {                 // K chunk of 32
        const int k0 = kc * 32;
        // ISA layout (16-bit A 16x32): lanes 0-15 hold K = 0..7 & 16..23,
        // lanes 16-31 hold K = 8..15 & 24..31 -> two contiguous 8-elem runs per lane.
        A16 a;
        #pragma unroll
        for (int j = 0; j < 8; ++j) {
            a.h[j]     = (_Float16)arow[k0 + j];
            a.h[j + 8] = (_Float16)arow[k0 + 16 + j];
        }
        #pragma unroll
        for (int nt = 0; nt < 8; ++nt) {             // 8 column tiles, reuse A
            const _Float16* brow = Bs + (nt * 16 + lm) * Dd + k0 + hs * 8;
            A16 b;
            #pragma unroll
            for (int j = 0; j < 8; ++j) {
                b.h[j]     = brow[j];
                b.h[j + 8] = brow[16 + j];
            }
            acc[nt].v = __builtin_amdgcn_wmma_f32_16x16x32_f16(
                /*neg_a=*/false, a.v, /*neg_b=*/false, b.v,
                /*c_mod=*/(short)0, acc[nt].v, /*reuse_a=*/false, /*reuse_b=*/false);
        }
    }
    // C/D layout: VGPR r -> M = r (lanes 0-15) or 8+r (lanes 16-31); N = lane%16
    #pragma unroll
    for (int nt = 0; nt < 8; ++nt) {
        #pragma unroll
        for (int r = 0; r < 8; ++r) {
            int row = mb + hs * 8 + r;
            XW[(size_t)row * Dd + nt * 16 + lm] = acc[nt].f[r];
        }
    }
}

// AGG = XW * snorm(i) + b   (self-loop term; also initializes the accumulator)
__global__ void k_init_agg(const float* __restrict__ XW, const float* __restrict__ dis,
                           const float* __restrict__ b, float* __restrict__ AGG) {
    size_t t = (size_t)blockIdx.x * blockDim.x + threadIdx.x;
    if (t >= (size_t)Nn * Dd) return;
    int i = (int)(t / Dd), d = (int)(t % Dd);
    float s = dis[i];
    AGG[t] = XW[t] * (s * s) + b[d];
}

// per-edge scatter: AGG[dst] += XW[src] * dis[src]*dis[dst]; 4 floats per thread
__global__ void k_edge_agg(const float* __restrict__ XW, const float* __restrict__ dis,
                           const int* __restrict__ src, const int* __restrict__ dst,
                           float* __restrict__ AGG) {
    int gid = blockIdx.x * blockDim.x + threadIdx.x;
    int e = gid >> 5, c = gid & 31;
    if (e >= Ee) return;
    int s = src[e], t = dst[e];
    float w = dis[s] * dis[t];
    const float* xs = XW + (size_t)s * Dd + c * 4;
    float* ag = AGG + (size_t)t * Dd + c * 4;
    atomicAdd(&ag[0], xs[0] * w);
    atomicAdd(&ag[1], xs[1] * w);
    atomicAdd(&ag[2], xs[2] * w);
    atomicAdd(&ag[3], xs[3] * w);
}

__global__ void k_leaky(const float* __restrict__ AGG, float* __restrict__ H) {
    size_t t = (size_t)blockIdx.x * blockDim.x + threadIdx.x;
    if (t < (size_t)Nn * Dd) H[t] = lrelu(AGG[t]);
}

// ---------------- BatchNorm over N rows (two pass, column atomics) ----------------
__global__ void __launch_bounds__(128)
k_bn_stats(const float* __restrict__ H, float* __restrict__ sums, float* __restrict__ sumsq) {
    int d  = threadIdx.x;               // column
    int r0 = blockIdx.x * 128;          // row chunk
    float s = 0.0f, ss = 0.0f;
    for (int r = 0; r < 128; ++r) {
        int row = r0 + r;
        if (row < Nn) {
            float v = H[(size_t)row * Dd + d];
            s += v; ss += v * v;
        }
    }
    atomicAdd(&sums[d], s);
    atomicAdd(&sumsq[d], ss);
}

__global__ void k_bn_apply(float* __restrict__ H, const float* __restrict__ sums,
                           const float* __restrict__ sumsq,
                           const float* __restrict__ gamma, const float* __restrict__ beta) {
    size_t t = (size_t)blockIdx.x * blockDim.x + threadIdx.x;
    if (t >= (size_t)Nn * Dd) return;
    int d = (int)(t % Dd);
    float m = sums[d] / (float)Nn;
    float v = sumsq[d] / (float)Nn - m * m;
    H[t] = (H[t] - m) * rsqrtf(v + EPSf) * gamma[d] + beta[d];
}

// ---------------- masked mean pools ----------------
__global__ void k_pool_cnt(const int* __restrict__ idx, const int* __restrict__ batch,
                           float* __restrict__ cnt) {
    int p = blockIdx.x * blockDim.x + threadIdx.x;
    if (p < NPp) atomicAdd(&cnt[batch[idx[p]]], 1.0f);
}

__global__ void k_pool_sum(const float* __restrict__ H, const int* __restrict__ idx,
                           const int* __restrict__ batch, float* __restrict__ pool) {
    int gid = blockIdx.x * blockDim.x + threadIdx.x;
    int p = gid >> 5, c = gid & 31;
    if (p >= NPp) return;
    int node = idx[p];
    int g = batch[node];
    const float* hr = H + (size_t)node * Dd + c * 4;
    float* pr = pool + (size_t)g * Dd + c * 4;
    atomicAdd(&pr[0], hr[0]);
    atomicAdd(&pr[1], hr[1]);
    atomicAdd(&pr[2], hr[2]);
    atomicAdd(&pr[3], hr[3]);
}

// comb = [pf, imf, pf-imf, pf*imf]  -> [G, 4D]
__global__ void k_comb(const float* __restrict__ pp, const float* __restrict__ pi,
                       const float* __restrict__ cp, const float* __restrict__ ci,
                       float* __restrict__ comb) {
    int t = blockIdx.x * blockDim.x + threadIdx.x;
    if (t >= Gg * Dd) return;
    int g = t / Dd, d = t % Dd;
    float pf  = pp[g * Dd + d] / fmaxf(cp[g], 1.0f);
    float imf = pi[g * Dd + d] / fmaxf(ci[g], 1.0f);
    float* row = comb + (size_t)g * (4 * Dd);
    row[d]           = pf;
    row[Dd + d]      = imf;
    row[2 * Dd + d]  = pf - imf;
    row[3 * Dd + d]  = pf * imf;
}

// ---------------- tiny head MLPs (scalar; ~10 MFLOP total) ----------------
__global__ void k_gf1(const float* __restrict__ comb, const float* __restrict__ w,
                      const float* __restrict__ b, float* __restrict__ h1) {
    int t = blockIdx.x * blockDim.x + threadIdx.x;          // G*D outputs
    if (t >= Gg * Dd) return;
    int g = t / Dd, j = t % Dd;
    float acc = b[j];
    const float* row = comb + (size_t)g * (4 * Dd);
    for (int k = 0; k < 4 * Dd; ++k) acc += row[k] * w[k * Dd + j];
    h1[t] = lrelu(acc);
}

__global__ void k_gf2(const float* __restrict__ h1, const float* __restrict__ w,
                      const float* __restrict__ b, float* __restrict__ h) {
    int t = blockIdx.x * blockDim.x + threadIdx.x;          // G*D outputs
    if (t >= Gg * Dd) return;
    int g = t / Dd, j = t % Dd;
    float acc = b[j];
    const float* row = h1 + (size_t)g * Dd;
    for (int k = 0; k < Dd; ++k) acc += row[k] * w[k * Dd + j];
    h[t] = acc;
}

// BN over G=128 rows, one block of 128 threads, one column each
__global__ void __launch_bounds__(128)
k_bn_small(float* __restrict__ h, const float* __restrict__ gamma, const float* __restrict__ beta) {
    int j = threadIdx.x;
    float s = 0.0f, ss = 0.0f;
    for (int g = 0; g < Gg; ++g) {
        float v = h[g * Dd + j];
        s += v; ss += v * v;
    }
    float m  = s / (float)Gg;
    float vv = ss / (float)Gg - m * m;
    float sc = rsqrtf(vv + EPSf) * gamma[j];
    for (int g = 0; g < Gg; ++g)
        h[g * Dd + j] = (h[g * Dd + j] - m) * sc + beta[j];
}

__global__ void k_fl1(const float* __restrict__ h, const float* __restrict__ w,
                      const float* __restrict__ b, float* __restrict__ o1) {
    int t = blockIdx.x * blockDim.x + threadIdx.x;          // G*64 outputs
    if (t >= Gg * 64) return;
    int g = t / 64, j = t % 64;
    float acc = b[j];
    const float* row = h + (size_t)g * Dd;
    for (int k = 0; k < Dd; ++k) acc += row[k] * w[k * 64 + j];
    o1[t] = lrelu(acc);
}

__global__ void k_fl2(const float* __restrict__ o1, const float* __restrict__ w,
                      const float* __restrict__ b, float* __restrict__ out) {
    int t = blockIdx.x * blockDim.x + threadIdx.x;          // G*32 outputs
    if (t >= Gg * 32) return;
    int g = t / 32, j = t % 32;
    float acc = b[j];
    const float* row = o1 + (size_t)g * 64;
    for (int k = 0; k < 64; ++k) acc += row[k] * w[k * 32 + j];
    out[t] = acc;
}

__global__ void k_out(const float* __restrict__ out, const float* __restrict__ w,
                      const float* __restrict__ b, float* __restrict__ prob) {
    int t = blockIdx.x * blockDim.x + threadIdx.x;          // G*2 outputs
    if (t >= Gg * 2) return;
    int g = t / 2, j = t % 2;
    float acc = b[j];
    const float* row = out + (size_t)g * 32;
    for (int k = 0; k < 32; ++k) acc += row[k] * w[k * 2 + j];
    prob[t] = acc;
}

// ---------------- host orchestration ----------------
extern "C" void kernel_launch(void* const* d_in, const int* in_sizes, int n_in,
                              void* d_out, int out_size, void* d_ws, size_t ws_size,
                              hipStream_t stream) {
    (void)in_sizes; (void)n_in; (void)out_size; (void)ws_size;

    const float* x         = (const float*)d_in[0];
    const float* W_gnn     = (const float*)d_in[1];
    const float* b_gnn     = (const float*)d_in[2];
    const float* hbn_gamma = (const float*)d_in[3];
    const float* hbn_beta  = (const float*)d_in[4];
    const float* gf_w1     = (const float*)d_in[5];
    const float* gf_b1     = (const float*)d_in[6];
    const float* gf_w2     = (const float*)d_in[7];
    const float* gf_b2     = (const float*)d_in[8];
    const float* bn2_gamma = (const float*)d_in[9];
    const float* bn2_beta  = (const float*)d_in[10];
    const float* fl_w1     = (const float*)d_in[11];
    const float* fl_b1     = (const float*)d_in[12];
    const float* fl_w2     = (const float*)d_in[13];
    const float* fl_b2     = (const float*)d_in[14];
    const float* out_w     = (const float*)d_in[15];
    const float* out_b     = (const float*)d_in[16];
    const int*   edge      = (const int*)d_in[17];   // [2,E]
    const int*   post_idx  = (const int*)d_in[18];
    const int*   image_idx = (const int*)d_in[19];
    const int*   batch_vec = (const int*)d_in[20];

    const int* e_src = edge;
    const int* e_dst = edge + Ee;

    float* outp  = (float*)d_out;            // out: G*32, then prob: G*2
    float* probp = outp + Gg * 32;

    // ---- workspace layout (floats) ----
    float* ws = (float*)d_ws;
    const size_t ND = (size_t)Nn * Dd;       // 6.4M
    float*    H     = ws;                    // N x D
    float*    XW    = ws + ND;               // N x D
    float*    AGG   = ws + 2 * ND;           // N x D
    float*    deg   = ws + 3 * ND;           // N
    float*    dis   = deg + Nn;              // N
    _Float16* W16T  = (_Float16*)(dis + Nn); // D*D halves (8192 floats)
    float*    sums  = (float*)W16T + 8192;   // D
    float*    sumsq = sums + Dd;             // D
    float*    pp    = sumsq + Dd;            // G x D  (post pool)
    float*    pi    = pp + Gg * Dd;          // G x D  (image pool)
    float*    cp    = pi + Gg * Dd;          // G
    float*    ci    = cp + Gg;               // G
    float*    comb  = ci + Gg;               // G x 4D
    float*    h1    = comb + Gg * 4 * Dd;    // G x D
    float*    hh    = h1 + Gg * Dd;          // G x D
    float*    o1    = hh + Gg * Dd;          // G x 64

    const int TPB = 256;
    const int ndBlocks = (int)((ND + TPB - 1) / TPB);

    // degree + norm
    k_zero<<<(Nn + TPB - 1) / TPB, TPB, 0, stream>>>(deg, Nn);
    k_deg<<<(Ee + TPB - 1) / TPB, TPB, 0, stream>>>(e_dst, deg);
    k_dis<<<(Nn + TPB - 1) / TPB, TPB, 0, stream>>>(deg, dis);

    // zero stats/pool region (sums..ci contiguous: 2D + 2GD + 2G = 33280)
    k_zero<<<(33280 + TPB - 1) / TPB, TPB, 0, stream>>>(sums, 33280);

    // ---- GNN layers ----
    const int rowTiles  = Nn / 16;                       // 3125
    const int gemmBlks  = (rowTiles + 7) / 8;            // 8 waves (row tiles) per block
    for (int l = 0; l < Ll; ++l) {
        const float* In = (l == 0) ? x : H;
        k_convert_w<<<(Dd * Dd + TPB - 1) / TPB, TPB, 0, stream>>>(W_gnn + (size_t)l * Dd * Dd, W16T);
        k_gemm<<<gemmBlks, 256, 0, stream>>>(In, W16T, XW);
        k_init_agg<<<ndBlocks, TPB, 0, stream>>>(XW, dis, b_gnn + (size_t)l * Dd, AGG);
        k_edge_agg<<<(Ee * 32) / TPB, TPB, 0, stream>>>(XW, dis, e_src, e_dst, AGG);
        k_leaky<<<ndBlocks, TPB, 0, stream>>>(AGG, H);
    }

    // ---- BatchNorm over nodes ----
    k_bn_stats<<<(Nn + 127) / 128, 128, 0, stream>>>(H, sums, sumsq);
    k_bn_apply<<<ndBlocks, TPB, 0, stream>>>(H, sums, sumsq, hbn_gamma, hbn_beta);

    // ---- mean pools ----
    k_pool_cnt<<<(NPp + TPB - 1) / TPB, TPB, 0, stream>>>(post_idx, batch_vec, cp);
    k_pool_cnt<<<(NPp + TPB - 1) / TPB, TPB, 0, stream>>>(image_idx, batch_vec, ci);
    k_pool_sum<<<(NPp * 32) / TPB, TPB, 0, stream>>>(H, post_idx, batch_vec, pp);
    k_pool_sum<<<(NPp * 32) / TPB, TPB, 0, stream>>>(H, image_idx, batch_vec, pi);
    k_comb<<<(Gg * Dd + TPB - 1) / TPB, TPB, 0, stream>>>(pp, pi, cp, ci, comb);

    // ---- head ----
    k_gf1<<<(Gg * Dd + TPB - 1) / TPB, TPB, 0, stream>>>(comb, gf_w1, gf_b1, h1);
    k_gf2<<<(Gg * Dd + TPB - 1) / TPB, TPB, 0, stream>>>(h1, gf_w2, gf_b2, hh);
    k_bn_small<<<1, 128, 0, stream>>>(hh, bn2_gamma, bn2_beta);
    k_fl1<<<(Gg * 64 + TPB - 1) / TPB, TPB, 0, stream>>>(hh, fl_w1, fl_b1, o1);
    k_fl2<<<(Gg * 32 + TPB - 1) / TPB, TPB, 0, stream>>>(o1, fl_w2, fl_b2, outp);
    k_out<<<(Gg * 2 + TPB - 1) / TPB, TPB, 0, stream>>>(outp, out_w, out_b, probp);
}
